// MessageFunction_36017595744927
// MI455X (gfx1250) — matrix operand order repaired and checked by reading, not compiled, not run on hardware
//
#include <hip/hip_runtime.h>
#include <hip/hip_bf16.h>

typedef __attribute__((ext_vector_type(16))) _Float16 v16h;
typedef __attribute__((ext_vector_type(8)))  float    v8f;

#define BATCH 8
#define NN    512
#define DIM   64
#define NCLS  4
#define NCH   (NN / 32)   // 16 K-chunks of 32
#define WTILE 32

// Total packed t elements per array (hi or lo): dirs * C * B * N * D
#define T_ELEMS ((size_t)2 * NCLS * BATCH * NN * DIM)   // 2,097,152 halfs = 4 MB

// ---------------------------------------------------------------------------
// Kernel 1: per-class pre-transform t[dir][c][b,w,d] = sum_e W[dir][c][d,e]*h[b,w,e]
// Split fp32 result into hi/lo f16 and scatter directly into the WMMA-B
// fragment layout (16x16x32 f16):  lane = (d&15) + 16*((k>>3)&1), half j = (k&7) + 8*((k&31)>>4).
// ---------------------------------------------------------------------------
__global__ __launch_bounds__(256) void gnn_pretransform(
    const float* __restrict__ h,     // [B,N,D]
    const float* __restrict__ Win,   // [C,D,D]
    const float* __restrict__ Wout,  // [C,D,D]
    _Float16* __restrict__ thi,
    _Float16* __restrict__ tlo)
{
    __shared__ float hlds[WTILE * DIM];   // 8 KB
    __shared__ float wlds[DIM * 65];      // transposed W, pitch 65 -> conflict-free

    const int wt = blockIdx.x;            // w tile: 0..15
    const int b  = blockIdx.y;            // 0..7
    const int t  = threadIdx.x;           // 0..255
    const int w0 = wt * WTILE;

    for (int i = t; i < WTILE * DIM; i += 256)
        hlds[i] = h[((size_t)b * NN + w0) * DIM + i];

    const int wl = t >> 3;                // local w row: 0..31
    const int d0 = t & 7;                 // d residue

    for (int dir = 0; dir < 2; ++dir) {
        const float* W = dir ? Wout : Win;
        for (int c = 0; c < NCLS; ++c) {
            __syncthreads();              // protects hlds first round + wlds reuse
            // load W[c] transposed: wlds[e*65+d] = W[c][d][e] (coalesced read,
            // stride-65 write -> consecutive banks)
            for (int i = t; i < DIM * DIM; i += 256) {
                const int d = i >> 6, e = i & 63;
                wlds[e * 65 + d] = W[(size_t)c * DIM * DIM + i];
            }
            __syncthreads();

            float acc[8] = {0.f,0.f,0.f,0.f,0.f,0.f,0.f,0.f};
            for (int e = 0; e < DIM; ++e) {
                const float hv = hlds[wl * DIM + e];   // broadcast across d group
                const float* wr = &wlds[e * 65 + d0];
                #pragma unroll
                for (int i = 0; i < 8; ++i)
                    acc[i] = fmaf(wr[8 * i], hv, acc[i]);
            }

            // scatter split halves into packed WMMA-B fragment layout
            const int wg   = w0 + wl;
            const int wc   = wg >> 5;          // chunk index (== wt)
            const int wk   = wg & 31;          // k within chunk
            const int lhi  = 16 * ((wk >> 3) & 1);   // lane high bit
            const int j    = (wk & 7) + ((wk >> 4) << 3);   // half index within fragment
            #pragma unroll
            for (int i = 0; i < 8; ++i) {
                const int d     = d0 + 8 * i;
                const int dtile = d >> 4;
                const int n     = d & 15;
                const size_t frag =
                    ((((size_t)dir * NCLS + c) * BATCH + b) * NCH + wc) * 4 + dtile;
                const size_t off = frag * 512 + (size_t)(lhi + n) * 16 + j;
                const float    v  = acc[i];
                const _Float16 hi = (_Float16)v;
                const _Float16 lo = (_Float16)(v - (float)hi);
                thi[off] = hi;
                tlo[off] = lo;
            }
        }
    }
}

// ---------------------------------------------------------------------------
// Kernel 2: msg[dir][b,v,d] = sum_c sum_w (onehot adj) * t[dir][c][b,w,d]
// One block per (b, 16-row v tile); 8 waves = {dir} x {d-tile}.
// A fragment = binary one-hot built from adj (exact in f16); B fragment =
// pre-swizzled hi/lo halves; two WMMAs accumulate into one f32 tile.
// ---------------------------------------------------------------------------
__global__ __launch_bounds__(256) void gnn_message(
    const int* __restrict__ adj,          // [B,N,N]
    const _Float16* __restrict__ thi,
    const _Float16* __restrict__ tlo,
    const float* __restrict__ bias,       // [2D]
    float* __restrict__ out)              // [B,N,2D]
{
    const int vt    = blockIdx.x;         // 0..31
    const int b     = blockIdx.y;         // 0..7
    const int lane  = threadIdx.x & 31;
    const int wid   = threadIdx.x >> 5;   // 0..7 (wave-uniform)
    const int dir   = wid >> 2;           // 0 = msg_in, 1 = msg_out
    const int dtile = wid & 3;            // 16-col tile of D
    const int vbase = vt * 16;
    const int m     = lane & 15;          // A-matrix row held by this lane
    const int kb    = (lane >> 4) * 8;    // A-layout k base (ISA 7.12.2)

    v8f acc = {};

    for (int wc = 0; wc < NCH; ++wc) {
        const int wbase = wc * 32;

        // gather the 16 adj values feeding this lane's A halfs
        // half j<8  -> k = kb + j ; half j>=8 -> k = kb + 16 + (j-8)
        int vals[16];
        if (dir == 0) {
            // rows of adj: fully coalesced int4 loads
            const int* row = adj + ((size_t)b * NN + vbase + m) * NN + wbase + kb;
            const int4 q0 = *(const int4*)(row);
            const int4 q1 = *(const int4*)(row + 4);
            const int4 q2 = *(const int4*)(row + 16);
            const int4 q3 = *(const int4*)(row + 20);
            vals[0]=q0.x;  vals[1]=q0.y;  vals[2]=q0.z;  vals[3]=q0.w;
            vals[4]=q1.x;  vals[5]=q1.y;  vals[6]=q1.z;  vals[7]=q1.w;
            vals[8]=q2.x;  vals[9]=q2.y;  vals[10]=q2.z; vals[11]=q2.w;
            vals[12]=q3.x; vals[13]=q3.y; vals[14]=q3.z; vals[15]=q3.w;
        } else {
            // transposed adj: per k, lanes 0..15 read 16 consecutive ints
            #pragma unroll
            for (int j = 0; j < 16; ++j) {
                const int k = kb + ((j < 8) ? j : (16 + (j - 8)));
                vals[j] = adj[((size_t)b * NN + wbase + k) * NN + vbase + m];
            }
        }

        #pragma unroll
        for (int c = 0; c < NCLS; ++c) {
            v16h a;
            #pragma unroll
            for (int j = 0; j < 16; ++j)
                a[j] = (vals[j] == c) ? (_Float16)1.0f : (_Float16)0.0f;

            const size_t frag =
                ((((size_t)dir * NCLS + c) * BATCH + b) * NCH + wc) * 4 + dtile;
            const v16h bh = *(const v16h*)(thi + frag * 512 + (size_t)lane * 16);
            const v16h bl = *(const v16h*)(tlo + frag * 512 + (size_t)lane * 16);

            acc = __builtin_amdgcn_wmma_f32_16x16x32_f16(
                      false, a, false, bh, (short)0, acc, false, false);
            acc = __builtin_amdgcn_wmma_f32_16x16x32_f16(
                      false, a, false, bl, (short)0, acc, false, false);
        }
    }

    // epilogue: C/D layout -> lane holds col n = lane&15, rows r + 8*(lane>>4)
    const int n     = lane & 15;
    const int mhalf = (lane >> 4) * 8;
    const int col   = dir * 64 + dtile * 16 + n;
    const float bv  = bias[col];
    #pragma unroll
    for (int r = 0; r < 8; ++r) {
        const int v = vbase + mhalf + r;
        out[((size_t)b * NN + v) * 128 + col] = acc[r] + bv;
    }
}

// ---------------------------------------------------------------------------
extern "C" void kernel_launch(void* const* d_in, const int* in_sizes, int n_in,
                              void* d_out, int out_size, void* d_ws, size_t ws_size,
                              hipStream_t stream) {
    const float* h    = (const float*)d_in[0];   // node_state [8,512,64]
    const int*   adj  = (const int*)  d_in[1];   // adj_mat    [8,512,512]
    const float* Win  = (const float*)d_in[2];   // matrix_in  [4,64,64]
    const float* Wout = (const float*)d_in[3];   // matrix_out [4,64,64]
    const float* bias = (const float*)d_in[4];   // bias       [128]
    float* out = (float*)d_out;                  // [8,512,128]

    _Float16* thi = (_Float16*)d_ws;             // 4 MB
    _Float16* tlo = thi + T_ELEMS;               // 4 MB  (needs 8 MB of ws)

    dim3 g1(NCH, BATCH);        // 16 x 8 blocks
    dim3 g2(NN / 16, BATCH);    // 32 x 8 blocks
    gnn_pretransform<<<g1, 256, 0, stream>>>(h, Win, Wout, thi, tlo);
    gnn_message<<<g2, 256, 0, stream>>>(adj, thi, tlo, bias, out);
}
